// MultiHeadAttention_24352464569715
// MI455X (gfx1250) — matrix operand order repaired and checked by reading, not compiled, not run on hardware
//
#include <hip/hip_runtime.h>
#include <hip/hip_bf16.h>
#include <math.h>

// ---------------------------------------------------------------------------
// Types for CDNA5 WMMA (wave32, 16x16x32 bf16 -> f32) and TDM descriptors
// ---------------------------------------------------------------------------
typedef __bf16 bf16;
typedef bf16  v16bf __attribute__((ext_vector_type(16)));
typedef bf16  v8bf  __attribute__((ext_vector_type(8)));
typedef float v8f   __attribute__((ext_vector_type(8)));

typedef unsigned int u32x4 __attribute__((ext_vector_type(4)));
typedef int          i32x8 __attribute__((ext_vector_type(8)));
typedef int          i32x4 __attribute__((ext_vector_type(4)));

#define WMMA_BF16(a, b, c) \
  __builtin_amdgcn_wmma_f32_16x16x32_bf16(false, (a), false, (b), (short)0, (c), false, false)

// Problem constants
#define BATCH 2
#define SEQ   2048
#define DMODEL 1024
#define NHEAD 16
#define DHEAD 64

// ---------------------------------------------------------------------------
// Tensor Data Mover: DMA one 2-D tile of 2-byte elements into LDS with row
// padding. Descriptor packing per CDNA5 ISA section 8.3/8.4.
//   pad_interval: code (pads after 2<<code DWORDs); pad_amount: code (+1 DWORDs)
// This toolchain exposes the 6-arg builtin:
//   (u32x4 g0, i32x8 g1, i32x4 g2, i32x4 g3, i32x8 extra, i32 cpol)
// ---------------------------------------------------------------------------
__device__ __forceinline__ void tdm_load_tile_2d(unsigned lds_addr, const void* gaddr,
                                                 unsigned tensor_d0, unsigned tensor_d1,
                                                 unsigned tile_d0, unsigned tile_d1,
                                                 unsigned stride0_elems,
                                                 unsigned pad_interval, unsigned pad_amount) {
  unsigned long long ga = (unsigned long long)(uintptr_t)gaddr;
  u32x4 g0;
  g0[0] = 1u;                                    // count=1, user descriptor
  g0[1] = lds_addr;                              // bits 63:32  lds_addr
  g0[2] = (unsigned)(ga & 0xffffffffu);          // bits 95:64  global_addr lo
  g0[3] = (unsigned)((ga >> 32) & 0x01ffffffu)   // bits 120:96 global_addr hi
        | (2u << 30);                            // bits 127:126 type=2 (image)
  i32x8 g1;
  g1[0] = (int)((1u << 16)                       // data_size = 2 bytes
              | (1u << 20)                       // pad_enable
              | (pad_interval << 22)
              | (pad_amount << 25));
  g1[1] = (int)((tensor_d0 & 0xffffu) << 16);    // tensor_dim0 lo16 (bits 63:48)
  g1[2] = (int)((tensor_d0 >> 16) | ((tensor_d1 & 0xffffu) << 16));
  g1[3] = (int)((tensor_d1 >> 16) | (tile_d0 << 16));
  g1[4] = (int)(tile_d1 & 0xffffu);              // tile_dim1; tile_dim2 = 0
  g1[5] = (int)stride0_elems;                    // tensor_dim0_stride lo32
  g1[6] = 0;                                     // stride0 hi16 / stride1 lo16
  g1[7] = 0;                                     // stride1 hi32
  i32x4 z4 = {};
  i32x8 z8 = {};
  __builtin_amdgcn_tensor_load_to_lds(g0, g1, z4, z4, z8, 0);
}

// ---------------------------------------------------------------------------
// f32 -> bf16 conversion kernels
// ---------------------------------------------------------------------------
__global__ __launch_bounds__(256) void cvt_f32_bf16(const float* __restrict__ in,
                                                    bf16* __restrict__ out, int n) {
  int i = blockIdx.x * 256 + threadIdx.x;
  if (i < n) out[i] = (bf16)in[i];
}

// in: [N,K] row-major (torch Linear weight) -> out: [K,N] row-major (W^T)
__global__ __launch_bounds__(256) void cvt_f32_bf16_T(const float* __restrict__ in,
                                                      bf16* __restrict__ out, int N, int K) {
  int i = blockIdx.x * 256 + threadIdx.x;
  if (i < N * K) {
    int n = i / K;
    int k = i - n * K;
    out[(size_t)k * N + n] = (bf16)in[i];
  }
}

// ---------------------------------------------------------------------------
// Tiled bf16 GEMM: C[M,N] = A[M,K] * Bt[K,N] + bias[N]
//   MODE 0: f32 output, row-major [M,N]
//   MODE 1: bf16 output, row-major [M,N]
//   MODE 2: bf16 output, transposed per-batch: out[(b*N + col)*S + s], row = b*S + s
// Workgroup: 256 threads (8 waves). Tile 128(M) x 128(N) x 32(K).
// Tiles staged into LDS by the Tensor Data Mover (wave 0 issues 2 descriptors
// per k-step, waits TENSORcnt, then the WG barrier releases all waves).
// ---------------------------------------------------------------------------
#define TILE_M 128
#define TILE_N 128
#define TILE_K 32
#define LDA (TILE_K + 8)   /* 40 bf16 -> 80B rows: 64B data + 16B TDM pad */
#define LDB (TILE_N + 8)   /* 136 bf16 -> 272B rows: 256B data + 16B TDM pad */

template <int MODE>
__global__ __launch_bounds__(256) void gemm_bf16(const bf16* __restrict__ A,
                                                 const bf16* __restrict__ Bt,
                                                 const float* __restrict__ bias,
                                                 void* __restrict__ Cout,
                                                 int M, int N, int K, int Sdim) {
  __shared__ __align__(16) bf16 Alds[TILE_M * LDA];
  __shared__ __align__(16) bf16 Blds[TILE_K * LDB];

  const int tid  = threadIdx.x;
  const int lane = tid & 31;
  const int wave = tid >> 5;
  const int m0 = blockIdx.x * TILE_M;
  const int n0 = blockIdx.y * TILE_N;

  const unsigned aLds = (unsigned)(uintptr_t)(void*)Alds;  // low 32 bits = LDS byte addr
  const unsigned bLds = (unsigned)(uintptr_t)(void*)Blds;

  v8f acc[8] = {};

  for (int kk = 0; kk < K; kk += TILE_K) {
    __syncthreads();   // previous k-step consumers done before DMA overwrites LDS
    if (tid < 32) {
      // A tile: 128 rows x 32 cols, row stride K; 64B rows + 16B pad (16DW interval, 4DW pad)
      tdm_load_tile_2d(aLds, A + (size_t)m0 * K + kk,
                       (unsigned)K, (unsigned)M, TILE_K, TILE_M, (unsigned)K,
                       /*pad_interval=*/3u, /*pad_amount=*/3u);
      // B tile: 32 rows x 128 cols of Bt[K,N]; 256B rows + 16B pad (64DW interval, 4DW pad)
      tdm_load_tile_2d(bLds, Bt + (size_t)kk * N + n0,
                       (unsigned)N, (unsigned)K, TILE_N, TILE_K, (unsigned)N,
                       /*pad_interval=*/5u, /*pad_amount=*/3u);
      __builtin_amdgcn_s_wait_tensorcnt(0);
    }
    __syncthreads();   // release all 8 waves onto the freshly-staged tiles

    // A operand: lane L<16 -> row 16w+L, K 0-7 & 16-23; lane L+16 -> same row, K 8-15 & 24-31
    v16bf a;
    {
      int row  = 16 * wave + (lane & 15);
      int koff = (lane >> 4) * 8;
      ((v8bf*)&a)[0] = *(const v8bf*)(Alds + row * LDA + koff);
      ((v8bf*)&a)[1] = *(const v8bf*)(Alds + row * LDA + koff + 16);
    }
#pragma unroll
    for (int nc = 0; nc < 8; ++nc) {
      // B operand: lane L -> N column L&15, K rows in table order (matches A)
      v16bf b;
#pragma unroll
      for (int s = 0; s < 16; ++s)
        b[s] = Blds[(((s & 8) << 1) + (lane >> 4) * 8 + (s & 7)) * LDB + nc * 16 + (lane & 15)];
      acc[nc] = WMMA_BF16(a, b, acc[nc]);
    }
  }

  // Epilogue. C/D layout: VGPR r, lanes 0-15 -> M=r, lanes 16-31 -> M=r+8; N=lane&15.
  const int rowbase  = m0 + 16 * wave + (lane >> 4) * 8;
  const int col_lane = lane & 15;
#pragma unroll
  for (int nc = 0; nc < 8; ++nc) {
    int col = n0 + nc * 16 + col_lane;
    float bv = bias ? bias[col] : 0.0f;
#pragma unroll
    for (int r = 0; r < 8; ++r) {
      int row = rowbase + r;
      float v = acc[nc][r] + bv;
      if (MODE == 0) {
        ((float*)Cout)[(size_t)row * N + col] = v;
      } else if (MODE == 1) {
        ((bf16*)Cout)[(size_t)row * N + col] = (bf16)v;
      } else {
        int bb = row / Sdim;
        int s  = row - bb * Sdim;
        ((bf16*)Cout)[((size_t)bb * N + col) * Sdim + s] = (bf16)v;
      }
    }
  }
}

// ---------------------------------------------------------------------------
// Flash attention: one wave per 16 query rows of one (batch, head).
//   Qp  : [B*S, D] bf16 (projected Q)
//   KpT : [B, D, S] bf16 (projected K, transposed)
//   Vp  : [B*S, D] bf16 (projected V)
//   Out : [B*S, D] bf16  (heads re-interleaved: col = h*64 + d)
// Streams KV in tiles of 32; online softmax; P bounced C-layout->A-layout via LDS.
// ---------------------------------------------------------------------------
__global__ __launch_bounds__(32) void attn_flash(const bf16* __restrict__ Qp,
                                                 const bf16* __restrict__ KpT,
                                                 const bf16* __restrict__ Vp,
                                                 const int* __restrict__ mask,
                                                 bf16* __restrict__ Out) {
  const int idx = blockIdx.x;          // ((b*16)+h)*128 + qt
  const int qt = idx & 127;
  const int h  = (idx >> 7) & 15;
  const int b  = idx >> 11;
  const int q0 = qt * 16;
  const int lane = threadIdx.x;

  __shared__ __align__(16) bf16 Plds[16 * 40];  // 16 x (32+8)

  // Q A-operands (row q0+(lane&15), d-chunks [0,32) and [32,64))
  v16bf aq0, aq1;
  {
    size_t base = (size_t)(b * SEQ + q0 + (lane & 15)) * DMODEL + h * DHEAD;
    int koff = (lane >> 4) * 8;
    ((v8bf*)&aq0)[0] = *(const v8bf*)(Qp + base + koff);
    ((v8bf*)&aq0)[1] = *(const v8bf*)(Qp + base + koff + 16);
    ((v8bf*)&aq1)[0] = *(const v8bf*)(Qp + base + 32 + koff);
    ((v8bf*)&aq1)[1] = *(const v8bf*)(Qp + base + 32 + koff + 16);
  }

  v8f o[4] = {};
  float mrow[8], lrow[8];
#pragma unroll
  for (int r = 0; r < 8; ++r) { mrow[r] = -__builtin_inff(); lrow[r] = 0.0f; }

  const float scale = 0.125f;  // 1/sqrt(64)

  for (int kv0 = 0; kv0 < SEQ; kv0 += 32) {
    // ---- scores: two 16x16 tiles (t = kv sub-tile)
    v8f s[2];
#pragma unroll
    for (int t = 0; t < 2; ++t) {
      v16bf bk0, bk1;  // B operand: lane L -> kv column L&15, d rows in table order (resp. +32)
      {
        size_t kc = (size_t)kv0 + t * 16 + (lane & 15);
#pragma unroll
        for (int s = 0; s < 16; ++s) {
          int k = ((s & 8) << 1) + (lane >> 4) * 8 + (s & 7);
          bk0[s] = KpT[((size_t)b * DMODEL + h * DHEAD + k) * SEQ + kc];
          bk1[s] = KpT[((size_t)b * DMODEL + h * DHEAD + 32 + k) * SEQ + kc];
        }
      }
      v8f c = {};
      c = WMMA_BF16(aq0, bk0, c);
      c = WMMA_BF16(aq1, bk1, c);
      int col = kv0 + t * 16 + (lane & 15);
      bool keep = mask[b * SEQ + col] != 0;
#pragma unroll
      for (int r = 0; r < 8; ++r) {
        float v = c[r] * scale;
        c[r] = keep ? v : -__builtin_inff();
      }
      s[t] = c;
    }

    // ---- online softmax (each row lives across 16 lanes of one acc element)
    float p0v[8], p1v[8];
#pragma unroll
    for (int r = 0; r < 8; ++r) {
      float t = fmaxf(s[0][r], s[1][r]);
#pragma unroll
      for (int g = 1; g < 16; g <<= 1) t = fmaxf(t, __shfl_xor(t, g, 32));
      float mnew = fmaxf(mrow[r], t);
      float sc = __expf(mrow[r] - mnew);
      p0v[r] = __expf(s[0][r] - mnew);
      p1v[r] = __expf(s[1][r] - mnew);
      float rs = p0v[r] + p1v[r];
#pragma unroll
      for (int g = 1; g < 16; g <<= 1) rs += __shfl_xor(rs, g, 32);
      lrow[r] = lrow[r] * sc + rs;
      mrow[r] = mnew;
#pragma unroll
      for (int j = 0; j < 4; ++j) o[j][r] *= sc;
    }

    // ---- bounce P through LDS: C-layout -> A-layout
    {
      int rbase = (lane >> 4) * 8;
      int c0 = lane & 15;
#pragma unroll
      for (int r = 0; r < 8; ++r) {
        Plds[(rbase + r) * 40 + c0]      = (bf16)p0v[r];
        Plds[(rbase + r) * 40 + 16 + c0] = (bf16)p1v[r];
      }
    }
    __syncthreads();  // single-wave WG: orders DS store->load via dscnt

    v16bf pa;
    {
      int row  = lane & 15;
      int koff = (lane >> 4) * 8;
      ((v8bf*)&pa)[0] = *(const v8bf*)(Plds + row * 40 + koff);
      ((v8bf*)&pa)[1] = *(const v8bf*)(Plds + row * 40 + koff + 16);
    }
    __syncthreads();

    // ---- P * V : B operand lane L -> d column L&15, kv rows in table order
#pragma unroll
    for (int j = 0; j < 4; ++j) {
      v16bf bv;
#pragma unroll
      for (int s = 0; s < 16; ++s)
        bv[s] = Vp[(size_t)(b * SEQ + kv0 + ((s & 8) << 1) + (lane >> 4) * 8 + (s & 7)) * DMODEL + h * DHEAD + j * 16 + (lane & 15)];
      o[j] = WMMA_BF16(pa, bv, o[j]);
    }
  }

  // ---- finalize: divide by l, store bf16 with heads interleaved back into [B*S, D]
  {
    int rbase = (lane >> 4) * 8;
    int c0 = lane & 15;
#pragma unroll
    for (int j = 0; j < 4; ++j) {
#pragma unroll
      for (int r = 0; r < 8; ++r) {
        float v = o[j][r] / lrow[r];
        size_t row = (size_t)(b * SEQ + q0 + rbase + r);
        Out[row * DMODEL + h * DHEAD + j * 16 + c0] = (bf16)v;
      }
    }
  }
}

// ---------------------------------------------------------------------------
// Host-side launcher
// ---------------------------------------------------------------------------
extern "C" void kernel_launch(void* const* d_in, const int* in_sizes, int n_in,
                              void* d_out, int out_size, void* d_ws, size_t ws_size,
                              hipStream_t stream) {
  (void)in_sizes; (void)n_in; (void)out_size; (void)ws_size;

  const float* Q    = (const float*)d_in[0];
  const float* K    = (const float*)d_in[1];
  const float* V    = (const float*)d_in[2];
  const int*   mask = (const int*)d_in[3];
  const float* Wq   = (const float*)d_in[4];
  const float* bq   = (const float*)d_in[5];
  const float* Wk   = (const float*)d_in[6];
  const float* bk   = (const float*)d_in[7];
  const float* Wv   = (const float*)d_in[8];
  const float* bv   = (const float*)d_in[9];
  const float* Wo   = (const float*)d_in[10];
  const float* bo   = (const float*)d_in[11];

  const int M = BATCH * SEQ;       // 4096
  const int N = DMODEL;            // 1024
  const int Kd = DMODEL;           // 1024
  const size_t actBytes = (size_t)M * DMODEL * sizeof(bf16);      // 8 MB
  const size_t wBytes   = (size_t)DMODEL * DMODEL * sizeof(bf16); // 2 MB

  char* ws = (char*)d_ws;
  bf16* Qbf  = (bf16*)(ws);
  bf16* Kbf  = (bf16*)(ws + actBytes);
  bf16* Vbf  = (bf16*)(ws + 2 * actBytes);
  bf16* WqT  = (bf16*)(ws + 3 * actBytes);
  bf16* WkT  = (bf16*)(ws + 3 * actBytes + wBytes);
  bf16* WvT  = (bf16*)(ws + 3 * actBytes + 2 * wBytes);
  bf16* WoT  = (bf16*)(ws + 3 * actBytes + 3 * wBytes);
  bf16* Qp   = (bf16*)(ws + 3 * actBytes + 4 * wBytes);
  bf16* KpT  = (bf16*)(ws + 4 * actBytes + 4 * wBytes);
  bf16* Vp   = (bf16*)(ws + 5 * actBytes + 4 * wBytes);
  bf16* Attn = (bf16*)(ws + 6 * actBytes + 4 * wBytes);

  const int nAct = M * DMODEL;           // 4,194,304
  const int nW   = DMODEL * DMODEL;      // 1,048,576

  // Stage 0: conversions
  cvt_f32_bf16<<<(nAct + 255) / 256, 256, 0, stream>>>(Q, Qbf, nAct);
  cvt_f32_bf16<<<(nAct + 255) / 256, 256, 0, stream>>>(K, Kbf, nAct);
  cvt_f32_bf16<<<(nAct + 255) / 256, 256, 0, stream>>>(V, Vbf, nAct);
  cvt_f32_bf16_T<<<(nW + 255) / 256, 256, 0, stream>>>(Wq, WqT, DMODEL, DMODEL);
  cvt_f32_bf16_T<<<(nW + 255) / 256, 256, 0, stream>>>(Wk, WkT, DMODEL, DMODEL);
  cvt_f32_bf16_T<<<(nW + 255) / 256, 256, 0, stream>>>(Wv, WvT, DMODEL, DMODEL);
  cvt_f32_bf16_T<<<(nW + 255) / 256, 256, 0, stream>>>(Wo, WoT, DMODEL, DMODEL);

  // Stage 1: projections
  dim3 ggrid(M / TILE_M, N / TILE_N);   // 32 x 8
  gemm_bf16<1><<<ggrid, 256, 0, stream>>>(Qbf, WqT, bq, Qp,  M, N, Kd, SEQ);
  gemm_bf16<2><<<ggrid, 256, 0, stream>>>(Kbf, WkT, bk, KpT, M, N, Kd, SEQ);
  gemm_bf16<1><<<ggrid, 256, 0, stream>>>(Vbf, WvT, bv, Vp,  M, N, Kd, SEQ);

  // Stage 2: flash attention (B*H*(S/16) waves)
  attn_flash<<<BATCH * NHEAD * (SEQ / 16), 32, 0, stream>>>(Qp, KpT, Vp, mask, Attn);

  // Stage 3: output projection -> f32 d_out
  gemm_bf16<0><<<ggrid, 256, 0, stream>>>(Attn, WoT, bo, (float*)d_out, M, N, Kd, SEQ);
}
